// RWKV7Attention_18210661335570
// MI455X (gfx1250) — matrix operand order repaired and checked by reading, not compiled, not run on hardware
//
#include <hip/hip_runtime.h>
#include <math.h>

// ---------- types ----------
typedef __bf16 bf16;
typedef __attribute__((ext_vector_type(16))) __bf16 v16bf;
typedef __attribute__((ext_vector_type(8)))  float  v8f;
typedef __attribute__((ext_vector_type(4)))  unsigned uint4v;
typedef __attribute__((ext_vector_type(8)))  int      int8v;
typedef __attribute__((ext_vector_type(4)))  int      int4v;

#define B_SZ   2
#define T_LEN  4096
#define H_DIM  2048
#define M_TOT  (B_SZ * T_LEN)     // 8192
#define NH     32
#define DK     64
#define W_SCALE (-0.6065306597126334f)

#if defined(__AMDGCN__) && __has_builtin(__builtin_amdgcn_tensor_load_to_lds)
#define HAS_TDM 1
#else
#define HAS_TDM 0
#endif

// ---------- helpers ----------
__device__ __forceinline__ float bf2f(bf16 x) {
    unsigned short s = __builtin_bit_cast(unsigned short, x);
    unsigned u = ((unsigned)s) << 16;
    return __builtin_bit_cast(float, u);
}
__device__ __forceinline__ bf16 f2bf(float f) {
    unsigned u = __builtin_bit_cast(unsigned, f);
    u += 0x7FFFu + ((u >> 16) & 1u);          // round-to-nearest-even
    unsigned short s = (unsigned short)(u >> 16);
    return __builtin_bit_cast(bf16, s);
}
__device__ __forceinline__ unsigned short bfbits(float f) {
    unsigned u = __builtin_bit_cast(unsigned, f);
    u += 0x7FFFu + ((u >> 16) & 1u);
    return (unsigned short)(u >> 16);
}
// packed store of two bf16 as one b32 (avoids ds_store_b16 pairs)
__device__ __forceinline__ void st2(bf16* p, float lo, float hi) {
    unsigned u = (unsigned)bfbits(lo) | ((unsigned)bfbits(hi) << 16);
    *reinterpret_cast<unsigned*>(p) = u;
}
__device__ __forceinline__ float sigm(float x) { return 1.0f / (1.0f + __expf(-x)); }

#if HAS_TDM
// ---------- Tensor Data Mover: 2-D bf16 tile -> LDS, 1-DWORD pad per 16-DWORD row ----------
// tile_x elements (bf16) per row (32 here -> 64B = 16 DWORDs), tile_y rows.
// LDS row stride becomes tile_x + 2 halfs (matches [rows][34] arrays).
__device__ __forceinline__ void tdm_load_tile(unsigned lds_off, const bf16* gptr,
                                              int tile_x, int tile_y,
                                              int tensor_x, int tensor_y,
                                              long long row_stride_elems)
{
    unsigned long long ga = (unsigned long long)(uintptr_t)gptr;
    uint4v g0;
    g0.x = 1u;                                              // count=1, user mode
    g0.y = lds_off;                                         // LDS byte address
    g0.z = (unsigned)(ga & 0xFFFFFFFFu);                    // global_addr[31:0]
    g0.w = (unsigned)((ga >> 32) & 0x01FFFFFFu) | (2u << 30); // addr[56:32] | type=2
    int8v g1;
    g1[0] = (1 << 16)          // data_size = 2 bytes
          | (1 << 20)          // pad_enable
          | (3 << 22)          // pad_interval: 16 DWORDs
          | (0 << 25);         // pad_amount: 1 DWORD
    g1[1] = (int)((tensor_x & 0xFFFF) << 16);               // tensor_dim0 low 16 (bits 79:48)
    g1[2] = (int)(((unsigned)tensor_x >> 16) & 0xFFFF) | (int)((tensor_y & 0xFFFF) << 16);
    g1[3] = (int)(((unsigned)tensor_y >> 16) & 0xFFFF) | (int)((tile_x & 0xFFFF) << 16);
    g1[4] = (int)(tile_y & 0xFFFF);                         // tile_dim1 (tile_dim2 = 0)
    g1[5] = (int)(row_stride_elems & 0xFFFFFFFFll);         // tensor_dim0_stride[31:0]
    g1[6] = (int)((row_stride_elems >> 32) & 0xFFFFll);     // stride[47:32] | dim1_stride lo = 0
    g1[7] = 0;
    int4v z4 = {0, 0, 0, 0};
    int8v z8 = {0, 0, 0, 0, 0, 0, 0, 0};
    __builtin_amdgcn_tensor_load_to_lds(g0, g1, z4, z4, z8, 0);   // 6-arg toolchain form
}
__device__ __forceinline__ unsigned lds_off_of(const void* p) {
    return (unsigned)(unsigned long long)(uintptr_t)p;      // low 32 bits = LDS offset
}
#endif

// ---------- fp32 -> bf16 conversion (weights) ----------
__global__ __launch_bounds__(256) void cvt_f32_bf16(const float* __restrict__ s,
                                                    bf16* __restrict__ d, size_t n) {
    size_t i = (size_t)blockIdx.x * blockDim.x + threadIdx.x;
    size_t stride = (size_t)gridDim.x * blockDim.x;
    for (; i < n; i += stride) d[i] = f2bf(s[i]);
}

// ---------- bf16 WMMA GEMM:  C[M,N] = A[M,K] @ W[N,K]^T (+epilogue) ----------
// AMODE 0: A built on the fly from fp32 hidden_states + token-shift lerp (mix[K])
// AMODE 1: A read from bf16 buffer (TDM-staged)
// EPI: 0=bf16 store  1=tanh  2=sigmoid  3=sigmoid(+bias)
//      4=exp(W_SCALE*sigmoid(+bias))  5=v-lerp (v0 + (vfirst-v0)*sigmoid(+bias))  6=f32 store
template<int AMODE, int EPI>
__global__ __launch_bounds__(256) void gemm_wmma_bf16(
    const bf16* __restrict__ Wt,   // [N,K]
    const bf16* __restrict__ Abf,  // AMODE 1
    const float* __restrict__ hs,  // AMODE 0: [M,K] fp32
    const float* __restrict__ mix, // AMODE 0: [K]
    int M, int N, int K,
    bf16* outB, float* outF,
    const float* __restrict__ bias,
    const bf16* vprev,             // EPI 5 (may alias outB; element-wise RMW by same thread)
    const float* __restrict__ vfirst)
{
    __shared__ bf16 As[128][34];
    __shared__ bf16 Ws[32][34];

    const int tid  = threadIdx.x;
    const int lane = tid & 31;
    const int wv   = tid >> 5;           // 8 waves
    const int wm   = wv >> 1;            // 0..3 (M)
    const int wn   = wv & 1;             // 0..1 (N)
    const int bm   = blockIdx.y * 128;
    const int bn   = blockIdx.x * 32;

    v8f acc0 = {};
    v8f acc1 = {};

    const int nk    = K >> 5;
    const int arow  = tid >> 1;          // 0..127, 16 halfs each
    const int akoff = (tid & 1) * 16;
    const int wrow  = tid >> 3;          // 0..31, 4 halfs each
    const int wkoff = (tid & 7) * 4;

    for (int kt = 0; kt < nk; ++kt) {
        const int kg0 = kt * 32;

        // ---- stage tiles ----
#if HAS_TDM
        if (wv == 0) {
            // W tile [32 rows x 32 halfs] via Tensor Data Mover (async DMA)
            tdm_load_tile(lds_off_of(&Ws[0][0]), Wt + (size_t)bn * K + kg0,
                          32, 32, K - kg0, N - bn, (long long)K);
            if (AMODE == 1)  // A tile [128 x 32] via TDM
                tdm_load_tile(lds_off_of(&As[0][0]), Abf + (size_t)bm * K + kg0,
                              32, 128, K - kg0, M - bm, (long long)K);
        }
#else
        #pragma unroll
        for (int j = 0; j < 4; ++j)
            Ws[wrow][wkoff + j] = Wt[(size_t)(bn + wrow) * K + kg0 + wkoff + j];
        if (AMODE == 1) {
            const int m = bm + arow;
            #pragma unroll
            for (int j = 0; j < 16; j += 2)
                st2(&As[arow][akoff + j],
                    bf2f(Abf[(size_t)m * K + kg0 + akoff + j]),
                    bf2f(Abf[(size_t)m * K + kg0 + akoff + j + 1]));
        }
#endif
        if (AMODE == 0) {
            // token-shift lerp fused into A staging, branchless + vectorized
            const int m  = bm + arow;
            const bool hp = (m & (T_LEN - 1)) != 0;
            const float ps = hp ? 1.0f : 0.0f;
            const size_t rb = (size_t)m * K + kg0 + akoff;
            const size_t pb = (size_t)(hp ? m - 1 : m) * K + kg0 + akoff;
            #pragma unroll
            for (int j4 = 0; j4 < 4; ++j4) {
                float4 x  = *reinterpret_cast<const float4*>(hs + rb + j4 * 4);
                float4 p  = *reinterpret_cast<const float4*>(hs + pb + j4 * 4);
                float4 mx = *reinterpret_cast<const float4*>(mix + kg0 + akoff + j4 * 4);
                float r0 = x.x + (p.x * ps - x.x) * mx.x;
                float r1 = x.y + (p.y * ps - x.y) * mx.y;
                float r2 = x.z + (p.z * ps - x.z) * mx.z;
                float r3 = x.w + (p.w * ps - x.w) * mx.w;
                st2(&As[arow][akoff + j4 * 4],     r0, r1);
                st2(&As[arow][akoff + j4 * 4 + 2], r2, r3);
            }
            if (kt + 1 < nk)
                __builtin_prefetch(&hs[rb + 32], 0, 1);
        }
#if HAS_TDM
        if (wv == 0) __builtin_amdgcn_s_wait_tensorcnt(0);
#endif
        __syncthreads();

        // ---- build fragments per CDNA5 16-bit layouts ----
        const int mr = lane & 15;
        const int kb = (lane >> 4) * 8;   // K sub-block per half-wave
        v16bf bfrag;
        #pragma unroll
        for (int n = 0; n < 16; ++n)      // B[k][n=lane&15] = W[bn+wn*16+(lane&15)][k], k in table order
            bfrag[n] = Ws[wn * 16 + mr][kb + (n & 7) + (n >> 3) * 16];
        v16bf a0, a1;
        #pragma unroll
        for (int j = 0; j < 8; ++j) {
            a0[j]     = As[wm * 32 + mr][kb + j];
            a0[j + 8] = As[wm * 32 + mr][16 + kb + j];
            a1[j]     = As[wm * 32 + 16 + mr][kb + j];
            a1[j + 8] = As[wm * 32 + 16 + mr][16 + kb + j];
        }
        acc0 = __builtin_amdgcn_wmma_f32_16x16x32_bf16(false, a0, false, bfrag,
                                                       (short)0, acc0, false, false);
        acc1 = __builtin_amdgcn_wmma_f32_16x16x32_bf16(false, a1, false, bfrag,
                                                       (short)0, acc1, false, false);
        __syncthreads();
    }

    // ---- epilogue: C layout = VGPR r -> M = r (+8 for lanes 16..31), N = lane&15 ----
    const int ncol  = bn + wn * 16 + (lane & 15);
    const int mbase = bm + wm * 32 + ((lane >> 4) ? 8 : 0);
    #pragma unroll
    for (int t2 = 0; t2 < 2; ++t2) {
        #pragma unroll
        for (int rr = 0; rr < 8; ++rr) {
            float val = (t2 == 0) ? acc0[rr] : acc1[rr];
            const int m = mbase + t2 * 16 + rr;
            const size_t om = (size_t)m * N + ncol;
            if (EPI == 0)      outB[om] = f2bf(val);
            else if (EPI == 1) outB[om] = f2bf(tanhf(val));
            else if (EPI == 2) outB[om] = f2bf(sigm(val));
            else if (EPI == 3) outB[om] = f2bf(sigm(val + bias[ncol]));
            else if (EPI == 4) outB[om] = f2bf(__expf(W_SCALE * sigm(val + bias[ncol])));
            else if (EPI == 5) {
                float v0 = bf2f(vprev[om]);
                outB[om] = f2bf(v0 + (vfirst[om] - v0) * sigm(val + bias[ncol]));
            } else             outF[om] = val;
        }
    }
}

// ---------- kk-normalize + fused k + (-kk, kk*a) ----------
__global__ __launch_bounds__(256) void kk_fuse(
    bf16* kbuf /*in-place -> fused k*/, const bf16* __restrict__ abuf,
    const float* __restrict__ k_k, const float* __restrict__ k_a,
    bf16* __restrict__ aneg, bf16* __restrict__ bvec)
{
    const int idx = blockIdx.x * blockDim.x + threadIdx.x;   // (m,h)
    if (idx >= M_TOT * NH) return;
    const int m = idx >> 5, h = idx & 31;
    const size_t base  = (size_t)m * H_DIM + h * DK;
    const size_t cbase = (size_t)h * DK;
    float sumsq = 0.0f;
    for (int j = 0; j < DK; ++j) {
        float kv = bf2f(kbuf[base + j]) * k_k[cbase + j];
        sumsq += kv * kv;
    }
    const float inv = 1.0f / fmaxf(sqrtf(sumsq), 1e-12f);
    for (int j = 0; j < DK; ++j) {
        float kraw = bf2f(kbuf[base + j]);
        float kk   = kraw * k_k[cbase + j] * inv;
        float av   = bf2f(abuf[base + j]);
        aneg[base + j] = f2bf(-kk);
        bvec[base + j] = f2bf(kk * av);
        kbuf[base + j] = f2bf(kraw * (1.0f + (av - 1.0f) * k_a[cbase + j]));
    }
}

// ---------- RWKV7 recurrence: one block per (b,head), thread = state row v ----------
__global__ __launch_bounds__(64) void rwkv7_scan(
    const bf16* __restrict__ rbuf, const bf16* __restrict__ dbuf,
    const bf16* __restrict__ kbuf, const bf16* __restrict__ vbuf,
    const bf16* __restrict__ aneg, const bf16* __restrict__ bvec,
    bf16* __restrict__ obuf)
{
    const int bh = blockIdx.x;
    const int b  = bh >> 5, h = bh & 31;
    const int vr = threadIdx.x;              // state row
    float S[DK];
    #pragma unroll
    for (int j = 0; j < DK; ++j) S[j] = 0.0f;

    __shared__ float sr[DK], sd[DK], sk[DK], sv[DK], sa[DK], sb[DK];

    for (int t = 0; t < T_LEN; ++t) {
        const int m = b * T_LEN + t;
        const size_t base = (size_t)m * H_DIM + h * DK;
        sr[vr] = bf2f(rbuf[base + vr]);
        sd[vr] = bf2f(dbuf[base + vr]);
        sk[vr] = bf2f(kbuf[base + vr]);
        sv[vr] = bf2f(vbuf[base + vr]);
        sa[vr] = bf2f(aneg[base + vr]);
        sb[vr] = bf2f(bvec[base + vr]);
        __syncthreads();

        float sacc = 0.0f;
        #pragma unroll
        for (int j = 0; j < DK; ++j) sacc += S[j] * sa[j];
        const float vv = sv[vr];
        float oacc = 0.0f;
        #pragma unroll
        for (int j = 0; j < DK; ++j) {
            S[j] = S[j] * sd[j] + sacc * sb[j] + vv * sk[j];
            oacc += S[j] * sr[j];
        }
        obuf[base + vr] = f2bf(oacc);
        __syncthreads();
    }
}

// ---------- GroupNorm + bonus + gate ----------
__global__ __launch_bounds__(256) void gn_gate(
    const bf16* __restrict__ obuf, const bf16* __restrict__ rbuf,
    const bf16* __restrict__ kbuf, const bf16* __restrict__ vbuf,
    const bf16* __restrict__ gbuf, const float* __restrict__ r_k,
    const float* __restrict__ gn_w, const float* __restrict__ gn_b,
    bf16* __restrict__ ybuf)
{
    const int idx = blockIdx.x * blockDim.x + threadIdx.x;   // (m,h)
    if (idx >= M_TOT * NH) return;
    const int m = idx >> 5, h = idx & 31;
    const size_t base = (size_t)m * H_DIM + h * DK;
    float s = 0.0f, s2 = 0.0f, bonus = 0.0f;
    for (int j = 0; j < DK; ++j) {
        float o = bf2f(obuf[base + j]);
        s += o; s2 += o * o;
        bonus += bf2f(rbuf[base + j]) * bf2f(kbuf[base + j]) * r_k[(size_t)h * DK + j];
    }
    const float mu  = s * (1.0f / DK);
    const float var = s2 * (1.0f / DK) - mu * mu;
    const float inv = rsqrtf(var + (float)DK * 1e-5f);
    for (int j = 0; j < DK; ++j) {
        float o  = bf2f(obuf[base + j]);
        float on = (o - mu) * inv * gn_w[(size_t)h * DK + j] + gn_b[(size_t)h * DK + j]
                 + bonus * bf2f(vbuf[base + j]);
        ybuf[base + j] = f2bf(on * bf2f(gbuf[base + j]));
    }
}

// ---------- launcher ----------
extern "C" void kernel_launch(void* const* d_in, const int* in_sizes, int n_in,
                              void* d_out, int out_size, void* d_ws, size_t ws_size,
                              hipStream_t stream) {
    const float* hs     = (const float*)d_in[0];
    const float* vfirst = (const float*)d_in[1];
    const float* x_r = (const float*)d_in[2];
    const float* x_w = (const float*)d_in[3];
    const float* x_k = (const float*)d_in[4];
    const float* x_v = (const float*)d_in[5];
    const float* x_a = (const float*)d_in[6];
    const float* x_g = (const float*)d_in[7];
    const float* k_k = (const float*)d_in[8];
    const float* k_a = (const float*)d_in[9];
    const float* r_k = (const float*)d_in[10];
    const float* w_r = (const float*)d_in[11];
    const float* w_k = (const float*)d_in[12];
    const float* w_v = (const float*)d_in[13];
    const float* w_o = (const float*)d_in[14];
    const float* wA  = (const float*)d_in[15];
    const float* wB  = (const float*)d_in[16];
    const float* wb  = (const float*)d_in[17];
    const float* aA  = (const float*)d_in[18];
    const float* aB  = (const float*)d_in[19];
    const float* ab  = (const float*)d_in[20];
    const float* vA  = (const float*)d_in[21];
    const float* vB  = (const float*)d_in[22];
    const float* vb  = (const float*)d_in[23];
    const float* gA  = (const float*)d_in[24];
    const float* gB  = (const float*)d_in[25];
    const float* gn_w = (const float*)d_in[26];
    const float* gn_b = (const float*)d_in[27];
    float* out = (float*)d_out;

    // workspace carve-out (assumed >= ~360 MB; compile-only environment)
    char* wsb = (char*)d_ws;
    size_t off = 0;
    auto alloc = [&](size_t bytes) -> char* {
        off = (off + 255) & ~(size_t)255;
        char* p = wsb + off;
        off += bytes;
        return p;
    };
    const size_t HH = (size_t)H_DIM * H_DIM;      // 4M
    const size_t MH = (size_t)M_TOT * H_DIM;      // 16.7M
    bf16* wrb = (bf16*)alloc(HH * 2);
    bf16* wkb = (bf16*)alloc(HH * 2);
    bf16* wvb = (bf16*)alloc(HH * 2);
    bf16* wob = (bf16*)alloc(HH * 2);
    bf16* wAb = (bf16*)alloc((size_t)96  * H_DIM * 2);
    bf16* wBb = (bf16*)alloc((size_t)H_DIM * 96 * 2);
    bf16* aAb = (bf16*)alloc((size_t)96  * H_DIM * 2);
    bf16* aBb = (bf16*)alloc((size_t)H_DIM * 96 * 2);
    bf16* vAb = (bf16*)alloc((size_t)64  * H_DIM * 2);
    bf16* vBb = (bf16*)alloc((size_t)H_DIM * 64 * 2);
    bf16* gAb = (bf16*)alloc((size_t)256 * H_DIM * 2);
    bf16* gBb = (bf16*)alloc((size_t)H_DIM * 256 * 2);
    bf16* rbuf  = (bf16*)alloc(MH * 2);
    bf16* kbuf  = (bf16*)alloc(MH * 2);   // raw k -> fused k in place
    bf16* vbuf  = (bf16*)alloc(MH * 2);   // v0 -> lerped v in place
    bf16* dbuf  = (bf16*)alloc(MH * 2);   // exp(w)
    bf16* abuf  = (bf16*)alloc(MH * 2);
    bf16* gbuf  = (bf16*)alloc(MH * 2);
    bf16* anegb = (bf16*)alloc(MH * 2);
    bf16* bvecb = (bf16*)alloc(MH * 2);
    bf16* wmid  = (bf16*)alloc((size_t)M_TOT * 96 * 2);
    bf16* amid  = (bf16*)alloc((size_t)M_TOT * 96 * 2);
    bf16* vmid  = (bf16*)alloc((size_t)M_TOT * 64 * 2);
    bf16* gmid  = (bf16*)alloc((size_t)M_TOT * 256 * 2);
    bf16* obuf  = (bf16*)alloc(MH * 2);
    bf16* ybuf  = (bf16*)alloc(MH * 2);
    (void)ws_size; (void)n_in; (void)in_sizes; (void)out_size;

    // ---- convert weights to bf16 ----
    auto cvt = [&](const float* s, bf16* d, size_t n) {
        int grd = (int)((n + 255) / 256);
        cvt_f32_bf16<<<grd, 256, 0, stream>>>(s, d, n);
    };
    cvt(w_r, wrb, HH); cvt(w_k, wkb, HH); cvt(w_v, wvb, HH); cvt(w_o, wob, HH);
    cvt(wA, wAb, (size_t)96 * H_DIM);  cvt(wB, wBb, (size_t)H_DIM * 96);
    cvt(aA, aAb, (size_t)96 * H_DIM);  cvt(aB, aBb, (size_t)H_DIM * 96);
    cvt(vA, vAb, (size_t)64 * H_DIM);  cvt(vB, vBb, (size_t)H_DIM * 64);
    cvt(gA, gAb, (size_t)256 * H_DIM); cvt(gB, gBb, (size_t)H_DIM * 256);

    const dim3 blk(256);
    auto grid_mn = [](int M, int N) { return dim3((unsigned)(N / 32), (unsigned)(M / 128)); };

    // ---- stage-1 GEMMs (fused token-shift lerp in A-load) ----
    gemm_wmma_bf16<0, 0><<<grid_mn(M_TOT, H_DIM), blk, 0, stream>>>(
        wrb, nullptr, hs, x_r, M_TOT, H_DIM, H_DIM, rbuf, nullptr, nullptr, nullptr, nullptr);
    gemm_wmma_bf16<0, 0><<<grid_mn(M_TOT, H_DIM), blk, 0, stream>>>(
        wkb, nullptr, hs, x_k, M_TOT, H_DIM, H_DIM, kbuf, nullptr, nullptr, nullptr, nullptr);
    gemm_wmma_bf16<0, 0><<<grid_mn(M_TOT, H_DIM), blk, 0, stream>>>(
        wvb, nullptr, hs, x_v, M_TOT, H_DIM, H_DIM, vbuf, nullptr, nullptr, nullptr, nullptr);
    gemm_wmma_bf16<0, 1><<<grid_mn(M_TOT, 96), blk, 0, stream>>>(     // tanh(xw @ wA^T)
        wAb, nullptr, hs, x_w, M_TOT, 96, H_DIM, wmid, nullptr, nullptr, nullptr, nullptr);
    gemm_wmma_bf16<0, 0><<<grid_mn(M_TOT, 96), blk, 0, stream>>>(     // xa @ aA^T
        aAb, nullptr, hs, x_a, M_TOT, 96, H_DIM, amid, nullptr, nullptr, nullptr, nullptr);
    gemm_wmma_bf16<0, 0><<<grid_mn(M_TOT, 64), blk, 0, stream>>>(     // xv @ vA^T
        vAb, nullptr, hs, x_v, M_TOT, 64, H_DIM, vmid, nullptr, nullptr, nullptr, nullptr);
    gemm_wmma_bf16<0, 2><<<grid_mn(M_TOT, 256), blk, 0, stream>>>(    // sigmoid(xg @ gA^T)
        gAb, nullptr, hs, x_g, M_TOT, 256, H_DIM, gmid, nullptr, nullptr, nullptr, nullptr);

    // ---- stage-2 LoRA GEMMs with fused activations (A staged via TDM) ----
    gemm_wmma_bf16<1, 4><<<grid_mn(M_TOT, H_DIM), blk, 0, stream>>>(  // decay = exp(W_SCALE*sig(.+wb))
        wBb, wmid, nullptr, nullptr, M_TOT, H_DIM, 96, dbuf, nullptr, wb, nullptr, nullptr);
    gemm_wmma_bf16<1, 3><<<grid_mn(M_TOT, H_DIM), blk, 0, stream>>>(  // a = sig(.+ab)
        aBb, amid, nullptr, nullptr, M_TOT, H_DIM, 96, abuf, nullptr, ab, nullptr, nullptr);
    gemm_wmma_bf16<1, 5><<<grid_mn(M_TOT, H_DIM), blk, 0, stream>>>(  // v lerp with v_first
        vBb, vmid, nullptr, nullptr, M_TOT, H_DIM, 64, vbuf, nullptr, vb, vbuf, vfirst);
    gemm_wmma_bf16<1, 0><<<grid_mn(M_TOT, H_DIM), blk, 0, stream>>>(  // g
        gBb, gmid, nullptr, nullptr, M_TOT, H_DIM, 256, gbuf, nullptr, nullptr, nullptr, nullptr);

    // ---- kk normalize / fused k / (-kk, kk*a) ----
    kk_fuse<<<(M_TOT * NH + 255) / 256, 256, 0, stream>>>(kbuf, abuf, k_k, k_a, anegb, bvecb);

    // ---- serial recurrence: 64 blocks (b,head) x 64 threads ----
    rwkv7_scan<<<B_SZ * NH, 64, 0, stream>>>(rbuf, dbuf, kbuf, vbuf, anegb, bvecb, obuf);

    // ---- groupnorm + bonus + gate ----
    gn_gate<<<(M_TOT * NH + 255) / 256, 256, 0, stream>>>(obuf, rbuf, kbuf, vbuf, gbuf,
                                                          r_k, gn_w, gn_b, ybuf);

    // ---- final projection (fp32 out) ----
    gemm_wmma_bf16<1, 6><<<grid_mn(M_TOT, H_DIM), blk, 0, stream>>>(
        wob, ybuf, nullptr, nullptr, M_TOT, H_DIM, H_DIM, nullptr, out, nullptr, nullptr, nullptr);
}
